// RoughMatchingModel_43636867728033
// MI455X (gfx1250) — matrix operand order repaired and checked by reading, not compiled, not run on hardware
//
#include <hip/hip_runtime.h>
#include <hip/hip_fp16.h>

// ---------------- problem constants (match reference) ----------------
#define NNODES   1024
#define NPTS     30000
#define DFEAT    256
#define NHEADS   4
#define DHEAD    64
#define KNEIGH   64
#define NCORR    256
#define OTITERS  100

typedef __attribute__((ext_vector_type(16))) _Float16 v16h;
typedef __attribute__((ext_vector_type(4)))  _Float16 v4h;
typedef __attribute__((ext_vector_type(8)))  float    v8f;
typedef __attribute__((ext_vector_type(4)))  float    v4f;

// =====================================================================
// Generic batched GEMM via v_wmma_f32_16x16x32_f16.
//   C[b] = alpha * op(A[b]) * op(B[b]) (+ bias[row]) (+ resid[b])
// op(A) is M x K, op(B) is K x N.  TA/TB are compile-time transpose flags.
// Block: 128 threads = 4 wave32 waves. Block tile 64x64, K-step 32.
// REQUIRES: M, N, Kd multiples of the tile and 16B-aligned pointers
// (all call sites use 64/256/1024 dims and 16-float-aligned buffers).
// Staging: two-phase -- 8x global_load_b128 issued back-to-back into
// registers, then f32->f16 convert + LDS stores in WMMA fragment order
// so each lane's 16-half fragment is a contiguous 32B run
// (2x ds_load_b128 per fragment in the MAC phase).
// =====================================================================
template <int TA, int TB>
__global__ __launch_bounds__(128)
void k_gemm_wmma(const float* __restrict__ A, const float* __restrict__ B,
                 const float* __restrict__ bias, const float* __restrict__ resid,
                 float* __restrict__ C,
                 int M, int N, int Kd,
                 int lda, int ldb, int ldc,
                 long long sA, long long sB, long long sC,
                 float alpha)
{
    // 4 wave-tiles (A) / 4 N-subtiles (B), each 32 lanes x 16 halfs
    __shared__ __align__(32) _Float16 lsA[4 * 512];
    __shared__ __align__(32) _Float16 lsB[4 * 512];

    const int bz = blockIdx.z;
    A += (long long)bz * sA;
    B += (long long)bz * sB;
    C += (long long)bz * sC;
    const float* R = resid ? (resid + (long long)bz * sC) : nullptr;

    const int m0   = blockIdx.y * 64;
    const int n0   = blockIdx.x * 64;
    const int tid  = threadIdx.x;
    const int lane = tid & 31;
    const int wave = tid >> 5;

    v8f acc[4] = {};

    for (int k0 = 0; k0 < Kd; k0 += 32) {
        const float* Ab = TA ? (A + (long long)k0 * lda + m0)
                             : (A + (long long)m0 * lda + k0);
        const float* Bb = TB ? (B + (long long)n0 * ldb + k0)
                             : (B + (long long)k0 * ldb + n0);

        // ---------- phase 1: issue all 8 vector loads ----------
        v4f ta[4], tb[4];
#pragma unroll
        for (int it = 0; it < 4; ++it) {
            int s = tid + it * 128;                 // 512 float4 slots
            if (TA == 0) {                          // contiguous along c (k)
                int r = s >> 3, c = (s & 7) * 4;
                ta[it] = *reinterpret_cast<const v4f*>(&Ab[(long long)r * lda + c]);
            } else {                                // contiguous along r (m)
                int c = s >> 4, r = (s & 15) * 4;
                ta[it] = *reinterpret_cast<const v4f*>(&Ab[(long long)c * lda + r]);
            }
        }
#pragma unroll
        for (int it = 0; it < 4; ++it) {
            int s = tid + it * 128;
            if (TB == 0) {                          // contiguous along c (n)
                int r = s >> 4, c = (s & 15) * 4;
                tb[it] = *reinterpret_cast<const v4f*>(&Bb[(long long)r * ldb + c]);
            } else {                                // contiguous along r (k)
                int c = s >> 3, r = (s & 7) * 4;
                tb[it] = *reinterpret_cast<const v4f*>(&Bb[(long long)c * ldb + r]);
            }
        }
        if (k0 + 32 < Kd) {                         // global_prefetch_b8
            __builtin_prefetch(TA ? (Ab + 32 * (long long)lda) : (Ab + 32), 0, 1);
            __builtin_prefetch(TB ? (Bb + 32) : (Bb + 32 * (long long)ldb), 0, 1);
        }

        // ---------- phase 2: convert + store in fragment order ----------
        // A reader: lane holds row (lane&15), kbA=(lane<16?0:8);
        //           e<8 -> k=kbA+e ; e>=8 -> k=16+kbA+(e-8)
#pragma unroll
        for (int it = 0; it < 4; ++it) {
            int s = tid + it * 128;
            v4h hv = __builtin_convertvector(ta[it], v4h);
            if (TA == 0) {
                int r = s >> 3, c = (s & 7) * 4;    // hi,e-run constant/consecutive
                int w  = r >> 4;
                int ln = (r & 15) + 16 * ((c >> 3) & 1);
                int e  = (c & 16) ? (8 + (c & 7)) : (c & 7);
                *reinterpret_cast<v4h*>(&lsA[w * 512 + ln * 16 + e]) = hv;
            } else {
                int c = s >> 4, r = (s & 15) * 4;
                int w  = r >> 4;
                int hi = (c >> 3) & 1;
                int e  = (c & 16) ? (8 + (c & 7)) : (c & 7);
#pragma unroll
                for (int q = 0; q < 4; ++q)
                    lsA[w * 512 + (((r + q) & 15) + 16 * hi) * 16 + e] = hv[q];
            }
        }
        // B reader: lane holds col (lane&15) of subtile t, kbB=(lane<16?0:8);
        //           e<8 -> k=kbB+e ; e>=8 -> k=16+kbB+(e-8)
#pragma unroll
        for (int it = 0; it < 4; ++it) {
            int s = tid + it * 128;
            v4h hv = __builtin_convertvector(tb[it], v4h);
            if (TB == 0) {
                int r = s >> 4, c = (s & 15) * 4;
                int t  = c >> 4;
                int ln = (c & 15) + 16 * ((r >> 3) & 1);
                int e  = (r & 16) ? (8 + (r & 7)) : (r & 7);
#pragma unroll
                for (int q = 0; q < 4; ++q)
                    lsB[t * 512 + (ln + q) * 16 + e] = hv[q];
            } else {
                int c = s >> 3, r = (s & 7) * 4;    // e-run consecutive, aligned
                int t  = c >> 4;
                int ln = (c & 15) + 16 * ((r >> 3) & 1);
                *reinterpret_cast<v4h*>(&lsB[t * 512 + ln * 16 + ((r & 16) ? (8 + (r & 7)) : (r & 7))]) = hv;
            }
        }
        __syncthreads();

        // ---------- MAC phase: 32B fragment loads + 4 WMMAs ----------
        v16h afrag = *reinterpret_cast<const v16h*>(&lsA[(wave * 32 + lane) * 16]);
#pragma unroll
        for (int t = 0; t < 4; ++t) {
            v16h bfrag = *reinterpret_cast<const v16h*>(&lsB[(t * 32 + lane) * 16]);
            acc[t] = __builtin_amdgcn_wmma_f32_16x16x32_f16(
                false, afrag, false, bfrag, (short)0, acc[t], false, false);
        }
        __syncthreads();
    }

    // ---- epilogue: C/D 16x16 f32 layout ----
    const int rbase = m0 + wave * 16 + ((lane >= 16) ? 8 : 0);
    const int cbase = n0 + (lane & 15);
#pragma unroll
    for (int t = 0; t < 4; ++t) {
#pragma unroll
        for (int r = 0; r < 8; ++r) {
            int row = rbase + r;
            int col = cbase + t * 16;
            float v = alpha * acc[t][r];
            if (bias) v += bias[row];
            if (R)    v += R[(long long)row * ldc + col];
            C[(long long)row * ldc + col] = v;
        }
    }
}

// ---------------- row softmax (in place), one block per row ----------------
__global__ void k_softmax(float* __restrict__ X, int L)
{
    long long row = blockIdx.x;
    float* p = X + row * (long long)L;
    int tid = threadIdx.x, T = blockDim.x;
    __shared__ float red[256];

    float m = -3.0e38f;
    for (int j = tid; j < L; j += T) m = fmaxf(m, p[j]);
    red[tid] = m; __syncthreads();
    for (int s = T >> 1; s > 0; s >>= 1) {
        if (tid < s) red[tid] = fmaxf(red[tid], red[tid + s]);
        __syncthreads();
    }
    m = red[0]; __syncthreads();

    float sum = 0.f;
    for (int j = tid; j < L; j += T) sum += __expf(p[j] - m);
    red[tid] = sum; __syncthreads();
    for (int s = T >> 1; s > 0; s >>= 1) {
        if (tid < s) red[tid] += red[tid + s];
        __syncthreads();
    }
    float inv = 1.f / red[0];
    for (int j = tid; j < L; j += T) p[j] = __expf(p[j] - m) * inv;
}

// ---------------- KNN: nearest node per point ----------------
__global__ __launch_bounds__(256)
void k_knn(const float* __restrict__ pts, const float* __restrict__ nodes,
           int* __restrict__ ids, int P)
{
    __shared__ float nx[NNODES], ny[NNODES], nz[NNODES], nn[NNODES];
    for (int i = threadIdx.x; i < NNODES; i += 256) {
        float a = nodes[i * 3], b = nodes[i * 3 + 1], c = nodes[i * 3 + 2];
        nx[i] = a; ny[i] = b; nz[i] = c; nn[i] = a * a + b * b + c * c;
    }
    __syncthreads();
    int p = blockIdx.x * 256 + threadIdx.x;
    if (p >= P) return;
    float px = pts[p * 3], py = pts[p * 3 + 1], pz = pts[p * 3 + 2];
    float best = 3.0e38f; int bi = 0;
    for (int i = 0; i < NNODES; ++i) {
        float d = -2.f * (px * nx[i] + py * ny[i] + pz * nz[i]) + nn[i];
        if (d < best) { best = d; bi = i; }   // first-min wins (argmin semantics)
    }
    ids[p] = bi;
}

// ---------------- group: first K point indices (ascending) per node ----------------
__global__ void k_group(const int* __restrict__ ids, int* __restrict__ patch, int P)
{
    int node = blockIdx.x * blockDim.x + threadIdx.x;
    if (node >= NNODES) return;
    int cnt = 0;
    for (int p = 0; p < P; ++p)
        if (ids[p] == node && cnt < KNEIGH) patch[node * KNEIGH + cnt++] = p;
    for (; cnt < KNEIGH; ++cnt) patch[node * KNEIGH + cnt] = -1;
}

// ---------------- gather features -> (C, D, K) transposed, plus mask ----------------
__global__ void k_gather(const float* __restrict__ ff, const int* __restrict__ patch,
                         const int* __restrict__ corr, int colsel,
                         float* __restrict__ out, int* __restrict__ maskOut)
{
    int c = blockIdx.x, k = blockIdx.y;
    int node = corr[c * 2 + colsel];
    int praw = patch[node * KNEIGH + k];
    int msk  = (praw < 0);
    int p    = msk ? 0 : praw;
    for (int d = threadIdx.x; d < DFEAT; d += blockDim.x)
        out[(long long)c * (DFEAT * KNEIGH) + (long long)d * KNEIGH + k] =
            ff[(long long)p * DFEAT + d];
    if (threadIdx.x == 0) maskOut[c * KNEIGH + k] = msk;
}

// ---------------- mask invalid local scores ----------------
__global__ void k_mask(float* __restrict__ lsc, const int* __restrict__ ms,
                       const int* __restrict__ mt)
{
    int i = blockIdx.x, c = blockIdx.y;
    int mi = ms[c * KNEIGH + i];
    float* row = lsc + (long long)c * (KNEIGH * KNEIGH) + (long long)i * KNEIGH;
    for (int j = threadIdx.x; j < KNEIGH; j += blockDim.x)
        if (mi || mt[c * KNEIGH + j]) row[j] = -10000.0f;
}

// ---------------- build padded Z with dustbin alpha ----------------
__global__ void k_buildZ(const float* __restrict__ S, const float* __restrict__ alpha_p,
                         float* __restrict__ Z, int m, int n, long long sS, long long sZ)
{
    int i = blockIdx.x, b = blockIdx.y;
    float a = alpha_p[0];
    float* Zr = Z + (long long)b * sZ + (long long)i * (n + 1);
    const float* Sr = S + (long long)b * sS + (long long)i * n;
    for (int j = threadIdx.x; j <= n; j += blockDim.x)
        Zr[j] = (i < m && j < n) ? Sr[j] : a;
}

__global__ void k_zero(float* p, int n)
{
    int i = blockIdx.x * blockDim.x + threadIdx.x;
    if (i < n) p[i] = 0.f;
}

// ---------------- Sinkhorn: combine two (max, scaled-sum) pairs ----------------
__device__ __forceinline__ void lse_combine(float& m1, float& s1, float m2, float s2)
{
    float M = fmaxf(m1, m2);
    s1 = s1 * __expf(m1 - M) + s2 * __expf(m2 - M);
    m1 = M;
}

// u[i] = log_mu[i] - logsumexp_j( Z[i,j] + v[j] );  one block per (row, batch)
__global__ void k_sink_u(const float* __restrict__ Z, const float* __restrict__ v,
                         float* __restrict__ u, int m, int n,
                         long long sZ, long long sU, long long sV)
{
    int i = blockIdx.x, b = blockIdx.y;
    const float* Zr = Z + (long long)b * sZ + (long long)i * (n + 1);
    const float* vv = v + (long long)b * sV;
    int tid = threadIdx.x, T = blockDim.x;
    float mx = -3.0e38f, s = 0.f;
    for (int j = tid; j <= n; j += T) {
        float x = Zr[j] + vv[j];
        if (x > mx) { s = s * __expf(mx - x) + 1.f; mx = x; }
        else        { s += __expf(x - mx); }
    }
    __shared__ float rm[256], rs[256];
    rm[tid] = mx; rs[tid] = s; __syncthreads();
    for (int st = T >> 1; st > 0; st >>= 1) {
        if (tid < st) lse_combine(rm[tid], rs[tid], rm[tid + st], rs[tid + st]);
        __syncthreads();
    }
    if (tid == 0) {
        float norm  = -__logf((float)(m + n));
        float logmu = (i < m) ? norm : (__logf((float)n) + norm);
        u[(long long)b * sU + i] = logmu - (rm[0] + __logf(rs[0]));
    }
}

// v[j] = log_nu[j] - logsumexp_i( Z[i,j] + u[i] );  one block per (col, batch)
__global__ void k_sink_v(const float* __restrict__ Z, const float* __restrict__ u,
                         float* __restrict__ v, int m, int n,
                         long long sZ, long long sU, long long sV)
{
    int j = blockIdx.x, b = blockIdx.y;
    const float* Zb = Z + (long long)b * sZ;
    const float* uu = u + (long long)b * sU;
    int tid = threadIdx.x, T = blockDim.x;
    float mx = -3.0e38f, s = 0.f;
    for (int i = tid; i <= m; i += T) {
        float x = Zb[(long long)i * (n + 1) + j] + uu[i];
        if (x > mx) { s = s * __expf(mx - x) + 1.f; mx = x; }
        else        { s += __expf(x - mx); }
    }
    __shared__ float rm[256], rs[256];
    rm[tid] = mx; rs[tid] = s; __syncthreads();
    for (int st = T >> 1; st > 0; st >>= 1) {
        if (tid < st) lse_combine(rm[tid], rs[tid], rm[tid + st], rs[tid + st]);
        __syncthreads();
    }
    if (tid == 0) {
        float norm  = -__logf((float)(m + n));
        float lognu = (j < n) ? norm : (__logf((float)m) + norm);
        v[(long long)b * sV + j] = lognu - (rm[0] + __logf(rs[0]));
    }
}

// out = Z + u[:,None] + v[None,:] - norm
__global__ void k_finish(const float* __restrict__ Z, const float* __restrict__ u,
                         const float* __restrict__ v, float* __restrict__ out,
                         int m, int n, long long sZ, long long sU, long long sV,
                         long long sOut)
{
    int i = blockIdx.x, b = blockIdx.y;
    float norm = -__logf((float)(m + n));
    float ui = u[(long long)b * sU + i];
    const float* Zr = Z + (long long)b * sZ + (long long)i * (n + 1);
    float* Or = out + (long long)b * sOut + (long long)i * (n + 1);
    const float* vv = v + (long long)b * sV;
    for (int j = threadIdx.x; j <= n; j += blockDim.x)
        Or[j] = Zr[j] + ui + vv[j] - norm;
}

// =====================================================================
// Host orchestration
// =====================================================================
extern "C" void kernel_launch(void* const* d_in, const int* in_sizes, int n_in,
                              void* d_out, int out_size, void* d_ws, size_t ws_size,
                              hipStream_t stream)
{
    (void)in_sizes; (void)n_in; (void)out_size; (void)ws_size;

    const float* src_pcd     = (const float*)d_in[0];
    const float* tgt_pcd     = (const float*)d_in[1];
    const float* src_nodes   = (const float*)d_in[2];
    const float* tgt_nodes   = (const float*)d_in[3];
    const float* src_feats   = (const float*)d_in[4];
    const float* tgt_feats   = (const float*)d_in[5];
    const float* src_final_f = (const float*)d_in[6];
    const float* tgt_final_f = (const float*)d_in[7];
    const float* attn_w      = (const float*)d_in[8];
    const float* attn_b      = (const float*)d_in[9];
    const float* fp_w        = (const float*)d_in[10];
    const float* fp_b        = (const float*)d_in[11];
    const float* l_attn_w    = (const float*)d_in[12];
    const float* l_attn_b    = (const float*)d_in[13];
    const float* lfp_w       = (const float*)d_in[14];
    const float* lfp_b       = (const float*)d_in[15];
    const float* bin_score   = (const float*)d_in[16];
    const float* l_bin_score = (const float*)d_in[17];
    const int*   node_corr   = (const int*)d_in[18];

    float* out_g = (float*)d_out;                       // (1025,1025)
    float* out_l = out_g + 1025 * 1025;                 // (256,65,65)

    // ------- workspace arena (~195 MB); every block rounded to 16 floats
    // so float4 (16B) accesses in the GEMM stay aligned -------
    size_t off = 0;
    auto alloc = [&](size_t nf) {
        float* p = (float*)d_ws + off;
        off += (nf + 15) & ~(size_t)15;
        return p;
    };

    float* A0 = alloc(DFEAT * NNODES);   // 256x1024 feature buffers
    float* A1 = alloc(DFEAT * NNODES);
    float* A2 = alloc(DFEAT * NNODES);
    float* A3 = alloc(DFEAT * NNODES);
    float* Qb = alloc(DFEAT * NNODES);
    float* Kb = alloc(DFEAT * NNODES);
    float* Vb = alloc(DFEAT * NNODES);
    float* Ob = alloc(DFEAT * NNODES);
    float* S4 = alloc((size_t)NHEADS * NNODES * NNODES); // per-head attention
    float* SC = alloc((size_t)NNODES * NNODES);          // global scores
    float* ZG = alloc((size_t)1025 * 1025);
    float* UG = alloc(1056);
    float* VG = alloc(1056);
    float* SF  = alloc((size_t)NCORR * DFEAT * KNEIGH);  // local (C,D,K)
    float* TF  = alloc((size_t)NCORR * DFEAT * KNEIGH);
    float* SF2 = alloc((size_t)NCORR * DFEAT * KNEIGH);
    float* TF2 = alloc((size_t)NCORR * DFEAT * KNEIGH);
    float* QL  = alloc((size_t)NCORR * DFEAT * KNEIGH);
    float* KL  = alloc((size_t)NCORR * DFEAT * KNEIGH);
    float* VL  = alloc((size_t)NCORR * DFEAT * KNEIGH);
    float* OL  = alloc((size_t)NCORR * DFEAT * KNEIGH);
    float* SL  = alloc((size_t)NCORR * NHEADS * KNEIGH * KNEIGH);
    float* LSC = alloc((size_t)NCORR * KNEIGH * KNEIGH);
    float* ZL  = alloc((size_t)NCORR * 65 * 65);
    float* UL  = alloc(NCORR * 65);
    float* VLv = alloc(NCORR * 65);
    int* ids_s   = (int*)alloc(NPTS);
    int* ids_t   = (int*)alloc(NPTS);
    int* patch_s = (int*)alloc(NNODES * KNEIGH);
    int* patch_t = (int*)alloc(NNODES * KNEIGH);
    int* ms      = (int*)alloc(NCORR * KNEIGH);
    int* mt      = (int*)alloc(NCORR * KNEIGH);

    auto gemm = [&](const float* A, const float* B, const float* bias,
                    const float* resid, float* Cm,
                    int M, int N, int Kd, int lda, int ldb, int ldc,
                    long long sA, long long sB, long long sC,
                    int tA, int tB, float alpha, int batch) {
        dim3 g((N + 63) / 64, (M + 63) / 64, batch);
        if (tA == 0 && tB == 0)
            k_gemm_wmma<0, 0><<<g, dim3(128), 0, stream>>>(A, B, bias, resid, Cm,
                M, N, Kd, lda, ldb, ldc, sA, sB, sC, alpha);
        else if (tA == 1 && tB == 0)
            k_gemm_wmma<1, 0><<<g, dim3(128), 0, stream>>>(A, B, bias, resid, Cm,
                M, N, Kd, lda, ldb, ldc, sA, sB, sC, alpha);
        else if (tA == 0 && tB == 1)
            k_gemm_wmma<0, 1><<<g, dim3(128), 0, stream>>>(A, B, bias, resid, Cm,
                M, N, Kd, lda, ldb, ldc, sA, sB, sC, alpha);
        else
            k_gemm_wmma<1, 1><<<g, dim3(128), 0, stream>>>(A, B, bias, resid, Cm,
                M, N, Kd, lda, ldb, ldc, sA, sB, sC, alpha);
    };

    // ---------------- global MHA: x,y are (256,1024) ----------------
    auto mhaG = [&](const float* x, const float* y, const float* W,
                    const float* Bv, float* dst) {
        gemm(W,          x, Bv,       nullptr, Qb, 256, 1024, 256, 256, 1024, 1024, 0, 0, 0, 0, 0, 1.f, 1);
        gemm(W + 65536,  y, Bv + 256, nullptr, Kb, 256, 1024, 256, 256, 1024, 1024, 0, 0, 0, 0, 0, 1.f, 1);
        gemm(W + 131072, y, Bv + 512, nullptr, Vb, 256, 1024, 256, 256, 1024, 1024, 0, 0, 0, 0, 0, 1.f, 1);
        // S[h][n,m] = (q_h^T k_h)/8 : batched over 4 heads
        gemm(Qb, Kb, nullptr, nullptr, S4, 1024, 1024, 64, 1024, 1024, 1024,
             65536, 65536, 1048576, 1, 0, 0.125f, NHEADS);
        k_softmax<<<dim3(NHEADS * 1024), dim3(256), 0, stream>>>(S4, 1024);
        // o[h][d,n] = sum_m S[h][n,m] v_h[d,m]
        gemm(Vb, S4, nullptr, nullptr, Ob, 64, 1024, 1024, 1024, 1024, 1024,
             65536, 1048576, 65536, 0, 1, 1.f, NHEADS);
        gemm(W + 196608, Ob, Bv + 768, x, dst, 256, 1024, 256, 256, 1024, 1024, 0, 0, 0, 0, 0, 1.f, 1);
    };

    // ---------------- local MHA: x,y are (C=256, 256, 64) ----------------
    auto mhaL = [&](const float* x, const float* y, const float* W,
                    const float* Bv, float* dst) {
        gemm(W,          x, Bv,       nullptr, QL, 256, 64, 256, 256, 64, 64, 0, 16384, 16384, 0, 0, 1.f, NCORR);
        gemm(W + 65536,  y, Bv + 256, nullptr, KL, 256, 64, 256, 256, 64, 64, 0, 16384, 16384, 0, 0, 1.f, NCORR);
        gemm(W + 131072, y, Bv + 512, nullptr, VL, 256, 64, 256, 256, 64, 64, 0, 16384, 16384, 0, 0, 1.f, NCORR);
        // batch = C*H = 1024, head stride 64*64=4096 tiles c-major/h-minor
        gemm(QL, KL, nullptr, nullptr, SL, 64, 64, 64, 64, 64, 64,
             4096, 4096, 4096, 1, 0, 0.125f, NCORR * NHEADS);
        k_softmax<<<dim3(NCORR * NHEADS * 64), dim3(64), 0, stream>>>(SL, 64);
        gemm(VL, SL, nullptr, nullptr, OL, 64, 64, 64, 64, 64, 64,
             4096, 4096, 4096, 0, 1, 1.f, NCORR * NHEADS);
        gemm(W + 196608, OL, Bv + 768, x, dst, 256, 64, 256, 256, 64, 64, 0, 16384, 16384, 0, 0, 1.f, NCORR);
    };

    // =========================== GLOBAL BRANCH ===========================
    mhaG(src_feats, src_feats, attn_w,          attn_b,       A0);  // L0 self
    mhaG(tgt_feats, tgt_feats, attn_w,          attn_b,       A1);
    mhaG(A0, A1, attn_w + 262144, attn_b + 1024, A2);               // L1 cross
    mhaG(A1, A0, attn_w + 262144, attn_b + 1024, A3);
    mhaG(A2, A2, attn_w + 524288, attn_b + 2048, A0);               // L2 self
    mhaG(A3, A3, attn_w + 524288, attn_b + 2048, A1);
    gemm(fp_w, A0, fp_b, nullptr, A2, 256, 1024, 256, 256, 1024, 1024, 0, 0, 0, 0, 0, 1.f, 1);
    gemm(fp_w, A1, fp_b, nullptr, A3, 256, 1024, 256, 256, 1024, 1024, 0, 0, 0, 0, 0, 1.f, 1);
    // scores[n,m] = sx^T tx / 16
    gemm(A2, A3, nullptr, nullptr, SC, 1024, 1024, 256, 1024, 1024, 1024, 0, 0, 0, 1, 0, 0.0625f, 1);

    k_buildZ<<<dim3(1025, 1), dim3(256), 0, stream>>>(SC, bin_score, ZG, 1024, 1024, 0, 0);
    k_zero<<<dim3((1025 + 255) / 256), dim3(256), 0, stream>>>(UG, 1025);
    k_zero<<<dim3((1025 + 255) / 256), dim3(256), 0, stream>>>(VG, 1025);
    for (int it = 0; it < OTITERS; ++it) {
        k_sink_u<<<dim3(1025, 1), dim3(256), 0, stream>>>(ZG, VG, UG, 1024, 1024, 0, 0, 0);
        k_sink_v<<<dim3(1025, 1), dim3(256), 0, stream>>>(ZG, UG, VG, 1024, 1024, 0, 0, 0);
    }
    k_finish<<<dim3(1025, 1), dim3(256), 0, stream>>>(ZG, UG, VG, out_g, 1024, 1024, 0, 0, 0, 0);

    // =========================== LOCAL BRANCH ===========================
    k_knn<<<dim3((NPTS + 255) / 256), dim3(256), 0, stream>>>(src_pcd, src_nodes, ids_s, NPTS);
    k_knn<<<dim3((NPTS + 255) / 256), dim3(256), 0, stream>>>(tgt_pcd, tgt_nodes, ids_t, NPTS);
    k_group<<<dim3(NNODES / 256), dim3(256), 0, stream>>>(ids_s, patch_s, NPTS);
    k_group<<<dim3(NNODES / 256), dim3(256), 0, stream>>>(ids_t, patch_t, NPTS);
    k_gather<<<dim3(NCORR, KNEIGH), dim3(256), 0, stream>>>(src_final_f, patch_s, node_corr, 0, SF, ms);
    k_gather<<<dim3(NCORR, KNEIGH), dim3(256), 0, stream>>>(tgt_final_f, patch_t, node_corr, 1, TF, mt);

    mhaL(SF, SF, l_attn_w,          l_attn_b,        SF2);   // L0 self
    mhaL(TF, TF, l_attn_w,          l_attn_b,        TF2);
    mhaL(SF2, TF2, l_attn_w + 262144, l_attn_b + 1024, SF);  // L1 cross
    mhaL(TF2, SF2, l_attn_w + 262144, l_attn_b + 1024, TF);
    mhaL(SF, SF, l_attn_w + 524288, l_attn_b + 2048, SF2);   // L2 self
    mhaL(TF, TF, l_attn_w + 524288, l_attn_b + 2048, TF2);
    gemm(lfp_w, SF2, lfp_b, nullptr, SF, 256, 64, 256, 256, 64, 64, 0, 16384, 16384, 0, 0, 1.f, NCORR);
    gemm(lfp_w, TF2, lfp_b, nullptr, TF, 256, 64, 256, 256, 64, 64, 0, 16384, 16384, 0, 0, 1.f, NCORR);
    // lsc[c][n,m] = sf^T tf / 16
    gemm(SF, TF, nullptr, nullptr, LSC, 64, 64, 256, 64, 64, 64,
         16384, 16384, 4096, 1, 0, 0.0625f, NCORR);
    k_mask<<<dim3(KNEIGH, NCORR), dim3(64), 0, stream>>>(LSC, ms, mt);

    k_buildZ<<<dim3(65, NCORR), dim3(64), 0, stream>>>(LSC, l_bin_score, ZL, 64, 64, 4096, 4225);
    k_zero<<<dim3((NCORR * 65 + 255) / 256), dim3(256), 0, stream>>>(UL, NCORR * 65);
    k_zero<<<dim3((NCORR * 65 + 255) / 256), dim3(256), 0, stream>>>(VLv, NCORR * 65);
    for (int it = 0; it < OTITERS; ++it) {
        k_sink_u<<<dim3(65, NCORR), dim3(64), 0, stream>>>(ZL, VLv, UL, 64, 64, 4225, 65, 65);
        k_sink_v<<<dim3(65, NCORR), dim3(64), 0, stream>>>(ZL, UL, VLv, 64, 64, 4225, 65, 65);
    }
    k_finish<<<dim3(65, NCORR), dim3(64), 0, stream>>>(ZL, UL, VLv, out_l, 64, 64, 4225, 65, 65, 4225);
}
